// MambaController_30253749633039
// MI455X (gfx1250) — matrix operand order repaired and checked by reading, not compiled, not run on hardware
//
#include <hip/hip_runtime.h>
#include <hip/hip_bf16.h>
#include <math.h>

// ---------------- problem constants ----------------
#define BATCH   8
#define SEQL    2048
#define MTOK    (BATCH * SEQL)      // 16384 tokens
#define OBS_DIM 256
#define DMODEL  768
#define DIN     1536                // D_INNER
#define DSTATE  16
#define DTRANK  48
#define DTRANKP 64                  // padded to multiple of 32 for WMMA K
#define XDBL_W  80                  // DTRANK + 2*DSTATE
#define XDBL_WP 128                 // xproj N padded to multiple of 64
#define DCONV   4
#define NLAYERS 2
#define EPSLN   1e-5f

typedef __attribute__((ext_vector_type(16))) _Float16 v16h;
typedef __attribute__((ext_vector_type(8)))  _Float16 v8h;
typedef __attribute__((ext_vector_type(8)))  float    v8f;

// Load a 16-half fragment: 8 halves at p, 8 halves at p+gap (two b128 loads).
__device__ __forceinline__ v16h ldfrag(const _Float16* __restrict__ p, int gap)
{
    v8h lo = *(const v8h*)p;
    v8h hi = *(const v8h*)(p + gap);
    v16h r;
#pragma unroll
    for (int i = 0; i < 8; ++i) { r[i] = lo[i]; r[8 + i] = hi[i]; }
    return r;
}

__device__ __forceinline__ v8f wmma_f16(v16h a, v16h b, v8f c)
{
    return __builtin_amdgcn_wmma_f32_16x16x32_f16(
        false, a, false, b, (short)0, c, false, false);
}

// ============================================================
// WMMA GEMM:  out[M,N] (ld=ldout) = A[M,K] (f16, ld=K) * W[Npad,K]^T (f16, ld=K)
// W must have >= gridDim.y*64 valid (possibly zero-padded) rows so B loads
// are unconditional.  ACT: 0 none, 1 softplus, 2 silu.  ADD: residual.
// Block = 128 threads = 4 waves; each wave computes a 32x64 tile:
// 2 A-fragments x 4 B-fragments -> 8 WMMAs per k-step, each B reused twice.
// ============================================================
template<typename OutT, int ACT, bool ADD, bool HAS_BIAS, int K>
__global__ __launch_bounds__(128)
void gemm_wmma_f16(const _Float16* __restrict__ A,
                   const _Float16* __restrict__ W,
                   OutT* __restrict__ out,
                   const float* __restrict__ bias,
                   int N, int ldout)
{
    const int lane   = threadIdx.x & 31;
    const int wave   = threadIdx.x >> 5;
    const int mBase  = blockIdx.x * 128 + wave * 32;
    const int nBase  = blockIdx.y * 64;
    const int r16    = lane & 15;
    const int hiHalf = lane >> 4;

    // A 16x32 f16, contiguous k order used here (not ISA table order): lanes 0-15 hold K 0-15, lanes 16-31 hold
    // K 16-31 (row M = lane%16).  B 32x16, same k order as A: lane = column, lanes 0-15
    // hold K 0-15, lanes 16-31 hold K 16-31, 2 consecutive K per VGPR.
    const int koffA = hiHalf ? 16 : 0;
    const int koffB = hiHalf ? 16 : 0;

    const _Float16* pa0 = A + (size_t)(mBase + r16) * K + koffA;
    const _Float16* pa1 = pa0 + (size_t)16 * K;
    const _Float16* pb0 = W + (size_t)(nBase +  0 + r16) * K + koffB;
    const _Float16* pb1 = W + (size_t)(nBase + 16 + r16) * K + koffB;
    const _Float16* pb2 = W + (size_t)(nBase + 32 + r16) * K + koffB;
    const _Float16* pb3 = W + (size_t)(nBase + 48 + r16) * K + koffB;

    v8f acc0[4], acc1[4];
#pragma unroll
    for (int t = 0; t < 4; ++t) { acc0[t] = {}; acc1[t] = {}; }

#pragma unroll 2
    for (int k0 = 0; k0 < K; k0 += 32) {
        const v16h a0 = ldfrag(pa0 + k0, 8);
        const v16h a1 = ldfrag(pa1 + k0, 8);
        v16h bf[4];
        bf[0] = ldfrag(pb0 + k0, 8);
        bf[1] = ldfrag(pb1 + k0, 8);
        bf[2] = ldfrag(pb2 + k0, 8);
        bf[3] = ldfrag(pb3 + k0, 8);
#pragma unroll
        for (int t = 0; t < 4; ++t) acc0[t] = wmma_f16(a0, bf[t], acc0[t]);
#pragma unroll
        for (int t = 0; t < 4; ++t) acc1[t] = wmma_f16(a1, bf[t], acc1[t]);
    }

    // D layout f32 16x16: lane l -> N = l%16 ; VGPR i -> M = i + (l>=16 ? 8:0)
    const int mrow0 = hiHalf * 8;
#pragma unroll
    for (int half = 0; half < 2; ++half) {
        const int mb = mBase + half * 16 + mrow0;
#pragma unroll
        for (int t = 0; t < 4; ++t) {
            const int n = nBase + t * 16 + r16;
            if (n >= N) continue;
            const v8f a = half ? acc1[t] : acc0[t];
#pragma unroll
            for (int i = 0; i < 8; ++i) {
                float v = a[i];
                if (HAS_BIAS) v += bias[n];
                if (ACT == 1) {      // softplus (numerically stable)
                    v = (v > 0.f) ? v + log1pf(__expf(-v)) : log1pf(__expf(v));
                } else if (ACT == 2) {  // silu
                    v = v / (1.f + __expf(-v));
                }
                const size_t idx = (size_t)(mb + i) * ldout + n;
                if (ADD) v += (float)out[idx];
                out[idx] = (OutT)v;
            }
        }
    }
}

// ============================================================
// LayerNorm over last dim D (wave32 per row), output f16
// ============================================================
__global__ __launch_bounds__(256)
void layernorm_f16(const float* __restrict__ x, const float* __restrict__ g,
                   const float* __restrict__ b, _Float16* __restrict__ out, int D)
{
    const int wave = threadIdx.x >> 5;
    const int lane = threadIdx.x & 31;
    const int row  = blockIdx.x * 8 + wave;
    const float* xr = x + (size_t)row * D;
    float s = 0.f, sq = 0.f;
    for (int j = lane; j < D; j += 32) { float v = xr[j]; s += v; sq += v * v; }
#pragma unroll
    for (int off = 16; off >= 1; off >>= 1) {
        s  += __shfl_xor(s,  off, 32);
        sq += __shfl_xor(sq, off, 32);
    }
    const float mean = s / D;
    const float var  = sq / D - mean * mean;
    const float inv  = rsqrtf(var + EPSLN);
    _Float16* orow = out + (size_t)row * D;
    for (int j = lane; j < D; j += 32)
        orow[j] = (_Float16)((xr[j] - mean) * inv * g[j] + b[j]);
}

// ============================================================
// Causal depthwise conv (taps=4) + silu; writes f32 and f16 copies
// ============================================================
__global__ __launch_bounds__(256)
void conv_silu(const float* __restrict__ u, const float* __restrict__ w,
               const float* __restrict__ cb, float* __restrict__ uc,
               _Float16* __restrict__ u16)
{
    const size_t idx = (size_t)blockIdx.x * 256 + threadIdx.x;  // over MTOK*DIN
    const int d  = (int)(idx % DIN);
    const size_t bt = idx / DIN;
    const int t  = (int)(bt % SEQL);
    const size_t brow = bt - t;               // b*SEQL
    float acc = cb[d];
#pragma unroll
    for (int i = 0; i < DCONV; ++i) {
        const int tt = t - (DCONV - 1) + i;
        if (tt >= 0) acc += w[d * DCONV + i] * u[(brow + tt) * DIN + d];
    }
    acc = acc / (1.f + __expf(-acc));          // silu
    uc[idx]  = acc;
    u16[idx] = (_Float16)acc;
}

// ============================================================
// Selective scan: one thread per (batch, channel); 16 states in regs.
// y written in-place over u.
// ============================================================
__global__ __launch_bounds__(256)
void selective_scan(float* __restrict__ u_y, const float* __restrict__ dt,
                    const float* __restrict__ xdbl,
                    const float* __restrict__ A_log,
                    const float* __restrict__ Dp)
{
    const int id = blockIdx.x * 256 + threadIdx.x;   // BATCH*DIN = 12288
    const int b = id / DIN;
    const int d = id - b * DIN;
    float Ar[DSTATE], h[DSTATE];
#pragma unroll
    for (int n = 0; n < DSTATE; ++n) {
        Ar[n] = -__expf(A_log[(size_t)d * DSTATE + n]);
        h[n]  = 0.f;
    }
    const float Dv = Dp[d];
    for (int t = 0; t < SEQL; ++t) {
        const size_t tok = (size_t)b * SEQL + t;
        const float uv  = u_y[tok * DIN + d];
        const float dtv = dt [tok * DIN + d];
        const float* xd = xdbl + tok * XDBL_W;
        float yv = 0.f;
#pragma unroll
        for (int n = 0; n < DSTATE; ++n) {
            const float dA = __expf(dtv * Ar[n]);
            h[n] = h[n] * dA + dtv * xd[DTRANK + n] * uv;
            yv  += h[n] * xd[DTRANK + DSTATE + n];
        }
        u_y[tok * DIN + d] = yv + uv * Dv;
    }
}

// ============================================================
// small elementwise helpers
// ============================================================
__global__ __launch_bounds__(256)
void mul_gate_f16(const float* __restrict__ y, const _Float16* __restrict__ zsilu,
                  _Float16* __restrict__ out)
{
    const size_t i = (size_t)blockIdx.x * 256 + threadIdx.x;
    out[i] = (_Float16)(y[i] * (float)zsilu[i]);
}

__global__ __launch_bounds__(256)
void cvt_f32_f16(const float* __restrict__ in, _Float16* __restrict__ out, size_t n)
{
    const size_t i = (size_t)blockIdx.x * 256 + threadIdx.x;
    if (i < n) out[i] = (_Float16)in[i];
}

// rows-padded f32 -> f16 conversion: out has rows_dst rows (extra rows zeroed)
__global__ __launch_bounds__(256)
void cvt_rows_pad(const float* __restrict__ in, _Float16* __restrict__ out,
                  int rows_src, int k, size_t total)
{
    const size_t i = (size_t)blockIdx.x * 256 + threadIdx.x;
    if (i >= total) return;
    const size_t row = i / k;
    const size_t col = i - row * k;
    out[i] = (row < (size_t)rows_src) ? (_Float16)in[row * k + col] : (_Float16)0.0f;
}

// dt_W (DIN x 48) -> f16 (DIN x 64, zero-padded K)
__global__ __launch_bounds__(256)
void cvt_dtw_pad(const float* __restrict__ in, _Float16* __restrict__ out)
{
    const size_t i = (size_t)blockIdx.x * 256 + threadIdx.x;  // DIN*64
    const int col = (int)(i & 63);
    const size_t row = i >> 6;
    out[i] = (col < DTRANK) ? (_Float16)in[row * DTRANK + col] : (_Float16)0.0f;
}

// x_dbl[:, :48] -> f16 (MTOK x 64, zero-padded K)
__global__ __launch_bounds__(256)
void pack_dtlr(const float* __restrict__ xdbl, _Float16* __restrict__ out)
{
    const size_t i = (size_t)blockIdx.x * 256 + threadIdx.x;  // MTOK*64
    const int col = (int)(i & 63);
    const size_t row = i >> 6;
    out[i] = (col < DTRANK) ? (_Float16)xdbl[row * XDBL_W + col] : (_Float16)0.0f;
}

// ============================================================
// launch
// ============================================================
extern "C" void kernel_launch(void* const* d_in, const int* in_sizes, int n_in,
                              void* d_out, int out_size, void* d_ws, size_t ws_size,
                              hipStream_t stream)
{
    const float* obs       = (const float*)d_in[0];
    const float* in_ng     = (const float*)d_in[1];
    const float* in_nb     = (const float*)d_in[2];
    const float* in_W      = (const float*)d_in[3];
    const float* in_b      = (const float*)d_in[4];
    const float* norm_g    = (const float*)d_in[5];
    const float* norm_b    = (const float*)d_in[6];
    const float* inproj_W  = (const float*)d_in[7];
    const float* conv_W    = (const float*)d_in[8];
    const float* conv_b    = (const float*)d_in[9];
    const float* xproj_W   = (const float*)d_in[10];
    const float* dt_W      = (const float*)d_in[11];
    const float* dt_b      = (const float*)d_in[12];
    const float* A_log     = (const float*)d_in[13];
    const float* D_param   = (const float*)d_in[14];
    const float* outproj_W = (const float*)d_in[15];
    const float* out_W     = (const float*)d_in[16];
    const float* out_b     = (const float*)d_in[17];
    float* out = (float*)d_out;

    // -------- workspace carve-up --------
    char* p = (char*)d_ws;
    auto take = [&](size_t bytes) { char* r = p; p += (bytes + 255) & ~(size_t)255; return r; };
    float*    X     = (float*)   take((size_t)MTOK * DMODEL * 4);   // residual stream
    _Float16* F16   = (_Float16*)take((size_t)MTOK * DIN * 2);      // xn / u16 / y16 staging
    float*    UDT   = (float*)   take((size_t)MTOK * DIN * 4);      // u (pre-conv), later dt
    float*    UC    = (float*)   take((size_t)MTOK * DIN * 4);      // conv(u); later y
    _Float16* Z16   = (_Float16*)take((size_t)MTOK * DIN * 2);      // silu(z)
    float*    XDBL  = (float*)   take((size_t)MTOK * XDBL_W * 4);
    _Float16* DTLR  = (_Float16*)take((size_t)MTOK * DTRANKP * 2);
    _Float16* wh_in   = (_Float16*)take((size_t)DMODEL * OBS_DIM * 2);
    _Float16* wh_inpj = (_Float16*)take((size_t)NLAYERS * 2 * DIN * DMODEL * 2);
    _Float16* wh_xpj  = (_Float16*)take((size_t)NLAYERS * XDBL_WP * DIN * 2); // N-padded
    _Float16* wh_dt   = (_Float16*)take((size_t)NLAYERS * DIN * DTRANKP * 2);
    _Float16* wh_opj  = (_Float16*)take((size_t)NLAYERS * DMODEL * DIN * 2);
    _Float16* wh_out  = (_Float16*)take((size_t)64 * DMODEL * 2);

    auto cvt = [&](const float* src, _Float16* dst, size_t n) {
        cvt_f32_f16<<<(unsigned)((n + 255) / 256), 256, 0, stream>>>(src, dst, n);
    };

    // -------- weight conversion (f32 -> f16 scratch) --------
    cvt(in_W,  wh_in,  (size_t)DMODEL * OBS_DIM);
    cvt(out_W, wh_out, (size_t)64 * DMODEL);
    for (int l = 0; l < NLAYERS; ++l) {
        cvt(inproj_W  + (size_t)l * 2 * DIN * DMODEL, wh_inpj + (size_t)l * 2 * DIN * DMODEL, (size_t)2 * DIN * DMODEL);
        cvt(outproj_W + (size_t)l * DMODEL * DIN,     wh_opj  + (size_t)l * DMODEL * DIN,     (size_t)DMODEL * DIN);
        const size_t xpj_total = (size_t)XDBL_WP * DIN;
        cvt_rows_pad<<<(unsigned)((xpj_total + 255) / 256), 256, 0, stream>>>(
            xproj_W + (size_t)l * XDBL_W * DIN, wh_xpj + (size_t)l * XDBL_WP * DIN,
            XDBL_W, DIN, xpj_total);
        cvt_dtw_pad<<<(DIN * DTRANKP) / 256, 256, 0, stream>>>(
            dt_W + (size_t)l * DIN * DTRANK, wh_dt + (size_t)l * DIN * DTRANKP);
    }

    const dim3 blk128(128), blk256(256);
    auto ggrid = [](int Npad) { return dim3(MTOK / 128, Npad / 64); };

    // -------- input stage: ln(obs) @ in_W^T + in_b --------
    layernorm_f16<<<MTOK / 8, blk256, 0, stream>>>(obs, in_ng, in_nb, F16, OBS_DIM);
    gemm_wmma_f16<float, 0, false, true, OBS_DIM><<<ggrid(DMODEL), blk128, 0, stream>>>(
        F16, wh_in, X, in_b, DMODEL, DMODEL);

    // -------- mamba layers --------
    for (int l = 0; l < NLAYERS; ++l) {
        const _Float16* Wi = wh_inpj + (size_t)l * 2 * DIN * DMODEL;

        layernorm_f16<<<MTOK / 8, blk256, 0, stream>>>(
            X, norm_g + l * DMODEL, norm_b + l * DMODEL, F16, DMODEL);

        // u = xn @ Wu^T ; silu(z) = silu(xn @ Wz^T) stored f16
        gemm_wmma_f16<float, 0, false, false, DMODEL><<<ggrid(DIN), blk128, 0, stream>>>(
            F16, Wi, UDT, nullptr, DIN, DIN);
        gemm_wmma_f16<_Float16, 2, false, false, DMODEL><<<ggrid(DIN), blk128, 0, stream>>>(
            F16, Wi + (size_t)DIN * DMODEL, Z16, nullptr, DIN, DIN);

        // depthwise causal conv + silu -> UC (f32) and F16 (f16 copy of u)
        conv_silu<<<(MTOK * DIN) / 256, blk256, 0, stream>>>(
            UDT, conv_W + (size_t)l * DIN * DCONV, conv_b + (size_t)l * DIN, UC, F16);

        // x_dbl = u @ xproj^T  (N = 80, weights padded to 128 rows)
        gemm_wmma_f16<float, 0, false, false, DIN><<<ggrid(XDBL_WP), blk128, 0, stream>>>(
            F16, wh_xpj + (size_t)l * XDBL_WP * DIN, XDBL, nullptr, XDBL_W, XDBL_W);

        // dt = softplus(dt_lr @ dt_W^T + dt_b)   (K padded 48->64) -> UDT (u is dead)
        pack_dtlr<<<(MTOK * DTRANKP) / 256, blk256, 0, stream>>>(XDBL, DTLR);
        gemm_wmma_f16<float, 1, false, true, DTRANKP><<<ggrid(DIN), blk128, 0, stream>>>(
            DTLR, wh_dt + (size_t)l * DIN * DTRANKP, UDT, dt_b + (size_t)l * DIN,
            DIN, DIN);

        // selective scan (y overwrites UC)
        selective_scan<<<(BATCH * DIN) / 256, blk256, 0, stream>>>(
            UC, UDT, XDBL, A_log + (size_t)l * DIN * DSTATE, D_param + (size_t)l * DIN);

        // y * silu(z) -> f16, then residual: X += y16 @ outproj^T
        mul_gate_f16<<<(MTOK * DIN) / 256, blk256, 0, stream>>>(UC, Z16, F16);
        gemm_wmma_f16<float, 0, true, false, DIN><<<ggrid(DMODEL), blk128, 0, stream>>>(
            F16, wh_opj + (size_t)l * DMODEL * DIN, X, nullptr, DMODEL, DMODEL);
    }

    // -------- head: out = X @ out_W^T + out_b --------
    cvt(X, F16, (size_t)MTOK * DMODEL);
    gemm_wmma_f16<float, 0, false, true, DMODEL><<<ggrid(64), blk128, 0, stream>>>(
        F16, wh_out, out, out_b, 64, 64);
}
